// DFFN_9904194585102
// MI455X (gfx1250) — compile-verified, matrix operand order checked by reading, not run on hardware
//
#include <hip/hip_runtime.h>
#include <hip/hip_bf16.h>
#include <math.h>

#define DIM   192
#define HID   510
#define HIDP  512
#define NB    8
#define HW    128

typedef __attribute__((ext_vector_type(16))) __bf16 v16bf;
typedef __attribute__((ext_vector_type(8)))  float  v8f;

union BF16Frag {
  v16bf  v;
  uint4  q[2];     // q[0] = elements 0..7, q[1] = elements 8..15
  __bf16 b[16];
};

// K index inside a 16-bit A/B WMMA fragment (ISA 7.12.2): element e, lane-group g
// e in 0..7  -> K = 8g + e        (contiguous 8 halfs)
// e in 8..15 -> K = 16 + 8g + e-8 (contiguous 8 halfs)
// => fragment = two aligned 16B chunks at K-offsets (8g) and (16+8g).

// ---------------------------------------------------------------------------
// Prep: bf16 weight packing (padded) + collapse 2x2-patch FFT filter into a
// per-channel real 4x4 transform T (PS=2 -> DFT matrix is Hadamard).
// ---------------------------------------------------------------------------
__global__ void prep_kernel(const float* __restrict__ w_in,
                            const float* __restrict__ w_out,
                            const float* __restrict__ fft_w,
                            __bf16* __restrict__ w_in_bf,
                            __bf16* __restrict__ w_out_bf,
                            float* __restrict__ T) {
  int tid = blockIdx.x * blockDim.x + threadIdx.x;
  if (tid < HIDP * DIM) {                    // w_in: [512][192], rows >=510 zero
    int m = tid / DIM, k = tid % DIM;
    w_in_bf[tid] = (m < HID) ? (__bf16)w_in[m * DIM + k] : (__bf16)0.f;
  }
  if (tid < DIM * HIDP) {                    // w_out: [192][512], cols >=510 zero
    int m = tid / HIDP, k = tid % HIDP;
    w_out_bf[tid] = (k < HID) ? (__bf16)w_out[m * HID + k] : (__bf16)0.f;
  }
  if (tid < HID * 16) {                      // T[c][i][j] = 1/4 sum_f w_f H[f][i] H[f][j]
    int c = tid >> 4, ij = tid & 15;
    int i = ij >> 2, j = ij & 3;
    int yi = i >> 1, xi = i & 1, yj = j >> 1, xj = j & 1;
    float s = 0.f;
#pragma unroll
    for (int f = 0; f < 4; ++f) {
      int u = f >> 1, v = f & 1;
      int sgn = ((u * yi + v * xi) + (u * yj + v * xj)) & 1;
      float w = fft_w[c * 4 + f];
      s += sgn ? -w : w;
    }
    T[tid] = 0.25f * s;
  }
}

// ---------------------------------------------------------------------------
// K1: h = FFTfilter( w_in @ x ). bf16 WMMA GEMM (M=512,K=192). N-tile = 2x8
// pixels so lanes {l, l^1, l^2, l^3} hold one 2x2 patch -> apply the 4x4
// patch transform on the accumulator with shfl_xor; store h as bf16.
// x tile staged through LDS ([pixel][channel], K-contiguous) so B fragments
// are two ds_load_b128 each; A fragments are two global_load_b128 each.
// ---------------------------------------------------------------------------
#define XS_STRIDE 208   // halfs; 208*2 = 416 bytes, 16B aligned rows

__global__ void __launch_bounds__(256)
k1_in_gemm_fft(const float* __restrict__ x,
               const __bf16* __restrict__ w_in_bf,
               const float* __restrict__ T,
               __bf16* __restrict__ h) {
  const int b    = blockIdx.y;
  const int tile = blockIdx.x;            // 0..1023
  const int ty   = tile >> 4;             // patch-row 0..63
  const int tx   = tile & 15;             // 0..15 (8 pixels each)
  const int t    = threadIdx.x;
  const int lane = t & 31;
  const int wave = t >> 5;
  const int n    = lane & 15;             // N column
  const int g    = lane >> 4;             // lane group
  const int i    = n & 3;                 // index inside 2x2 patch
  const int q    = n >> 2;                // patch within tile
  const int y    = ty * 2 + (i >> 1);
  const int xp   = tx * 8 + q * 2 + (i & 1);

  __shared__ __attribute__((aligned(16))) __bf16 xs[16 * XS_STRIDE];

  // ---- stage x tile (192 channels x 16 pixels) into LDS as bf16 ----
  {
    const int sn = t & 15;                // pixel this thread stages
    const int k0 = t >> 4;                // 0..15
    const int si = sn & 3, sq = sn >> 2;
    const int sy = ty * 2 + (si >> 1);
    const int sx = tx * 8 + sq * 2 + (si & 1);
    const float* xb = x + (((size_t)b * DIM) * HW + sy) * HW + sx;
#pragma unroll
    for (int j = 0; j < 12; ++j) {
      const int k = k0 + 16 * j;
      xs[sn * XS_STRIDE + k] = (__bf16)xb[(size_t)k * HW * HW];
    }
  }
  __syncthreads();

  // ---- B fragments: 2 x b128 from LDS per kblock ----
  BF16Frag bfrag[6];
#pragma unroll
  for (int kb = 0; kb < 6; ++kb) {
    const uint4* p = (const uint4*)(xs + n * XS_STRIDE + kb * 32 + 8 * g);
    bfrag[kb].q[0] = p[0];               // K = kb*32 + 8g .. +7
    bfrag[kb].q[1] = p[2];               // K = kb*32 + 16 + 8g .. +7
  }

#pragma unroll
  for (int mt = 0; mt < 4; ++mt) {
    const int mtile = wave * 4 + mt;      // 0..31 (512 rows / 16)
    const int row   = mtile * 16 + n;     // A rows ride lane%16
    v8f acc = {};
#pragma unroll
    for (int kb = 0; kb < 6; ++kb) {
      BF16Frag a;
      const uint4* ap = (const uint4*)(w_in_bf + row * DIM + kb * 32 + 8 * g);
      a.q[0] = ap[0];
      a.q[1] = ap[2];
      acc = __builtin_amdgcn_wmma_f32_16x16x32_bf16(false, a.v, false, bfrag[kb].v,
                                                    (short)0, acc, false, false);
    }
    // ---- per-channel 4x4 patch transform on the D fragment + bf16 store ----
#pragma unroll
    for (int r = 0; r < 8; ++r) {
      float p0 = acc[r];
      float p1 = __shfl_xor(p0, 1, 32);
      float p2 = __shfl_xor(p0, 2, 32);
      float p3 = __shfl_xor(p0, 3, 32);
      const int m = mtile * 16 + r + (g << 3);   // D row = r + 8*(lane>=16)
      if (m < HID) {
        const float* Tm = T + m * 16 + i * 4;
        float res = Tm[i] * p0 + Tm[i ^ 1] * p1 + Tm[i ^ 2] * p2 + Tm[i ^ 3] * p3;
        h[(((size_t)b * HID + m) * HW + y) * HW + xp] = (__bf16)res;
      }
    }
  }
}

// ---------------------------------------------------------------------------
// K2: out = w_out @ ( gelu(dw1(h)) * dw2(h) ). 128 threads build the gated
// B block (32 channels x 16 pixels) in LDS ([pixel][channel], K-contiguous),
// 4 waves run the 192x512 GEMM with bf16 WMMA, fp32 accumulate, fp32 store.
// ---------------------------------------------------------------------------
#define GS_STRIDE 48    // halfs; 48*2 = 96 bytes, 16B aligned rows

__global__ void __launch_bounds__(128)
k2_dw_gate_out(const __bf16* __restrict__ h,
               const float* __restrict__ w_dw,
               const __bf16* __restrict__ w_out_bf,
               float* __restrict__ out) {
  const int b    = blockIdx.z;
  const int y    = blockIdx.y;
  const int x0   = blockIdx.x * 16;
  const int t    = threadIdx.x;
  const int lane = t & 31;
  const int wave = t >> 5;
  const int n    = lane & 15;
  const int g    = lane >> 4;
  __shared__ __attribute__((aligned(16))) __bf16 g_lds[16 * GS_STRIDE];

  v8f acc0 = {}, acc1 = {}, acc2 = {};

  const int pn  = t & 15;                 // pixel this thread gates
  const int cl0 = t >> 4;                 // 0..7

  for (int kb = 0; kb < 16; ++kb) {       // K = 512 = 16 blocks of 32
    // prefetch next block's h rows (global_prefetch_b8)
    if (kb + 1 < 16) {
      const int cn = (kb + 1) * 32 + cl0;
      if (cn < HID) {
        __builtin_prefetch(h + ((size_t)b * HID + (cn >> 1)) * HW * HW + (size_t)y * HW + x0, 0, 0);
        __builtin_prefetch(h + ((size_t)b * HID + ((cn + HID) >> 1)) * HW * HW + (size_t)y * HW + x0, 0, 0);
      }
    }
    // ---- depthwise 3x3 + exact gelu gate into LDS ----
#pragma unroll
    for (int j = 0; j < 4; ++j) {
      const int clocal = cl0 + j * 8;     // 0..31
      const int c = kb * 32 + clocal;
      float gv = 0.f;
      if (c < HID) {
        const int c1 = c >> 1;                 // x1 source channel
        const int c2 = (c + HID) >> 1;         // x2 source channel
        const float* wd1 = w_dw + (size_t)c * 9;
        const float* wd2 = w_dw + (size_t)(c + HID) * 9;
        const size_t base1 = ((size_t)b * HID + c1) * HW * HW;
        const size_t base2 = ((size_t)b * HID + c2) * HW * HW;
        float x1 = 0.f, x2 = 0.f;
#pragma unroll
        for (int dy = -1; dy <= 1; ++dy) {
          const int yy = y + dy;
          if (yy < 0 || yy >= HW) continue;
#pragma unroll
          for (int dx = -1; dx <= 1; ++dx) {
            const int xx = x0 + pn + dx;
            if (xx < 0 || xx >= HW) continue;
            x1 += (float)h[base1 + (size_t)yy * HW + xx] * wd1[(dy + 1) * 3 + (dx + 1)];
            x2 += (float)h[base2 + (size_t)yy * HW + xx] * wd2[(dy + 1) * 3 + (dx + 1)];
          }
        }
        const float ge = 0.5f * x1 * (1.f + erff(x1 * 0.70710678118f));
        gv = ge * x2;
      }
      g_lds[pn * GS_STRIDE + clocal] = (__bf16)gv;
    }
    __syncthreads();

    // ---- B fragment: 2 x b128 from LDS ----
    BF16Frag bfrag;
    {
      const uint4* p = (const uint4*)(g_lds + n * GS_STRIDE + 8 * g);
      bfrag.q[0] = p[0];
      bfrag.q[1] = p[2];
    }

    // ---- 12 M-tiles across 4 waves (3 each) ----
#pragma unroll
    for (int tm = 0; tm < 3; ++tm) {
      const int mtile = wave + tm * 4;    // 0..11
      const int row   = mtile * 16 + n;
      BF16Frag a;
      const uint4* ap = (const uint4*)(w_out_bf + row * HIDP + kb * 32 + 8 * g);
      a.q[0] = ap[0];
      a.q[1] = ap[2];
      v8f& acc = (tm == 0) ? acc0 : ((tm == 1) ? acc1 : acc2);
      acc = __builtin_amdgcn_wmma_f32_16x16x32_bf16(false, a.v, false, bfrag.v,
                                                    (short)0, acc, false, false);
    }
    __syncthreads();
  }

  // ---- store fp32 output ----
#pragma unroll
  for (int tm = 0; tm < 3; ++tm) {
    const int mtile = wave + tm * 4;
    const v8f& acc = (tm == 0) ? acc0 : ((tm == 1) ? acc1 : acc2);
#pragma unroll
    for (int r = 0; r < 8; ++r) {
      const int m = mtile * 16 + r + (g << 3);
      out[(((size_t)b * DIM + m) * HW + y) * HW + x0 + n] = acc[r];
    }
  }
}

// ---------------------------------------------------------------------------
extern "C" void kernel_launch(void* const* d_in, const int* in_sizes, int n_in,
                              void* d_out, int out_size, void* d_ws, size_t ws_size,
                              hipStream_t stream) {
  const float* x     = (const float*)d_in[0];   // (8,192,128,128)
  const float* w_in  = (const float*)d_in[1];   // (510,192,1,1)
  const float* w_dw  = (const float*)d_in[2];   // (1020,1,3,3)
  const float* fft_w = (const float*)d_in[3];   // (510,1,1,2,2)
  const float* w_out = (const float*)d_in[4];   // (192,510,1,1)
  float* out = (float*)d_out;                   // (8,192,128,128)

  char* ws = (char*)d_ws;
  __bf16* w_in_bf  = (__bf16*)(ws);                  // 512*192*2   = 196608 B
  __bf16* w_out_bf = (__bf16*)(ws + 196608);         // 192*512*2   = 196608 B
  float*  T        = (float*)(ws + 393216);          // 510*16*4    =  32640 B
  __bf16* hbuf     = (__bf16*)(ws + 425984);         // 8*510*128*128*2 ~ 133 MB

  prep_kernel<<<(HIDP * DIM + 255) / 256, 256, 0, stream>>>(w_in, w_out, fft_w,
                                                            w_in_bf, w_out_bf, T);
  k1_in_gemm_fft<<<dim3(1024, NB), 256, 0, stream>>>(x, w_in_bf, T, hbuf);
  k2_dw_gate_out<<<dim3(8, HW, NB), 128, 0, stream>>>(hbuf, w_dw, w_out_bf, out);
}
